// NERModel_9131100471836
// MI455X (gfx1250) — compile-verified, matrix operand order checked by reading, not run on hardware
//
#include <hip/hip_runtime.h>
#include <hip/hip_bf16.h>

#define B_   64
#define S_   512
#define E_   256
#define H_   512
#define HD_  1024
#define T_   9
#define G4H  2048   // 4*H

typedef __attribute__((ext_vector_type(8)))  __bf16 v8bf;
typedef __attribute__((ext_vector_type(16))) __bf16 v16bf;
typedef __attribute__((ext_vector_type(8)))  float  v8f;

__device__ __forceinline__ unsigned short f2bf(float f) {
  union { float f; unsigned u; } v; v.f = f;
  unsigned u = v.u;
  unsigned r = 0x7FFFu + ((u >> 16) & 1u);   // round-to-nearest-even
  return (unsigned short)((u + r) >> 16);
}
__device__ __forceinline__ float bf2f(unsigned short u) {
  union { unsigned u; float f; } v; v.u = ((unsigned)u) << 16; return v.f;
}
__device__ __forceinline__ float sigmoidf_(float x) { return 1.0f / (1.0f + expf(-x)); }

__device__ __forceinline__ v16bf mk16(v8bf lo, v8bf hi) {
  v16bf r;
#pragma unroll
  for (int i = 0; i < 8; ++i) { r[i] = lo[i]; r[8 + i] = hi[i]; }
  return r;
}

// ---------------- grid-wide barrier (device-scope atomics, generation counter) ----
__device__ __forceinline__ void grid_barrier(unsigned* cnt, unsigned* gen, unsigned nblocks) {
  __syncthreads();
  if (threadIdx.x == 0) {
    unsigned g = __hip_atomic_load(gen, __ATOMIC_RELAXED, __HIP_MEMORY_SCOPE_AGENT);
    unsigned arrived = __hip_atomic_fetch_add(cnt, 1u, __ATOMIC_ACQ_REL, __HIP_MEMORY_SCOPE_AGENT) + 1u;
    if (arrived == nblocks) {
      __hip_atomic_store(cnt, 0u, __ATOMIC_RELAXED, __HIP_MEMORY_SCOPE_AGENT);
      __hip_atomic_fetch_add(gen, 1u, __ATOMIC_ACQ_REL, __HIP_MEMORY_SCOPE_AGENT);
    } else {
      while (__hip_atomic_load(gen, __ATOMIC_ACQUIRE, __HIP_MEMORY_SCOPE_AGENT) == g)
        __builtin_amdgcn_s_sleep(2);
    }
  }
  __syncthreads();
}

// ---------------- init: barrier state ------------------------------------------
__global__ void init_kernel(unsigned* cnt, unsigned* gen) {
  if (threadIdx.x == 0) { *cnt = 0u; *gen = 0u; }
}

// ---------------- embedding gather + f32->bf16 ----------------------------------
__global__ void embed_kernel(const int* __restrict__ ids, const float* __restrict__ emb,
                             unsigned short* __restrict__ x_bf) {
  size_t i = (size_t)blockIdx.x * blockDim.x + threadIdx.x;
  if (i >= (size_t)B_ * S_ * E_) return;
  int e = (int)(i & (E_ - 1));
  size_t row = i >> 8;            // E_ == 256
  int tok = ids[row];
  x_bf[i] = f2bf(emb[(size_t)tok * E_ + e]);
}

// ---------------- weight prepack --------------------------------------------------
// Wcat[dir][n'][k] bf16 with GATE-INTERLEAVED column permutation:
//   n' = g*16 + gate*4 + jj   <->   n_orig = gate*512 + g*4 + jj
// so one 16-wide WMMA N-tile carries {i,f,g,o} for 4 consecutive hidden units,
// letting the LSTM cell update happen entirely in-wave via lane shuffles.
__global__ void prepack_kernel(const float* __restrict__ w_ih, const float* __restrict__ w_hh,
                               const float* __restrict__ b_ih, const float* __restrict__ b_hh,
                               unsigned short* __restrict__ Wcat, float* __restrict__ bias, int Ein) {
  int Ktot = Ein + H_;
  size_t total  = (size_t)2 * G4H * Ktot;
  size_t stride = (size_t)gridDim.x * blockDim.x;
  for (size_t i = (size_t)blockIdx.x * blockDim.x + threadIdx.x; i < total; i += stride) {
    int k = (int)(i % Ktot);
    size_t nd = i / Ktot;
    int np = (int)(nd % G4H);          // permuted column
    int d  = (int)(nd / G4H);
    int g  = np >> 4;
    int c  = np & 15;
    int n  = (c >> 2) * H_ + (g << 2) + (c & 3);   // original gate column
    float v = (k < Ein) ? w_ih[((size_t)d * G4H + n) * Ein + k]
                        : w_hh[((size_t)d * G4H + n) * H_ + (k - Ein)];
    Wcat[i] = f2bf(v);
  }
  size_t gid = (size_t)blockIdx.x * blockDim.x + threadIdx.x;
  if (gid < 2 * G4H) {
    int d  = (int)(gid / G4H);
    int np = (int)(gid % G4H);
    int g  = np >> 4;
    int c  = np & 15;
    int n  = (c >> 2) * H_ + (g << 2) + (c & 3);
    bias[gid] = b_ih[(size_t)d * G4H + n] + b_hh[(size_t)d * G4H + n];
  }
}

// ---------------- persistent bidirectional LSTM layer ---------------------------
// grid = 64 blocks x 128 threads (4 waves). wave -> (dir, 4-hidden-unit j-block).
// Per step: fused GEMM  gates = [x_t | h_{t-1}] @ Wcat^T + bias  via WMMA bf16;
// cell state c lives in registers; i/f/g/o exchanged in-wave via shuffles;
// only the h broadcast crosses the grid -> ONE barrier per timestep.
__global__ __launch_bounds__(128) void lstm_layer_kernel(
    const unsigned short* __restrict__ x_in,   // [B][S][Ein] bf16
    const unsigned short* __restrict__ Wcat,   // [2][2048][Ktot] bf16 (permuted N x K)
    const float*          __restrict__ bias,   // [2][2048] (permuted)
    unsigned short*       __restrict__ out,    // [B][S][1024] bf16
    unsigned short*       __restrict__ h_bf,   // [2][64][512] bf16
    unsigned* cnt, unsigned* gen, int Ein)
{
  const int Ktot  = Ein + H_;
  const int tid   = threadIdx.x;
  const int wv    = tid >> 5;
  const int lane  = tid & 31;
  const int gw    = blockIdx.x * 4 + wv;   // 0..255
  const int dir   = gw >> 7;               // 0/1
  const int strip = gw & 127;              // 0..127  -> j-block [4*strip, 4*strip+4)
  const int n0    = strip * 16;
  const int mrow  = lane & 15;             // A: row-in-tile, B: column, C: N index
  const int hi8   = (lane >> 4) << 3;      // 0 or 8
  const unsigned gthreads = gridDim.x * blockDim.x;

  extern __shared__ __align__(16) char smem_raw[];
  __bf16* ldsB = (__bf16*)smem_raw;
  const int ldstr = Ktot + 8;              // pad 16B to avoid bank conflicts
  __bf16* myB = ldsB + (size_t)wv * 16 * ldstr;

  // Stage this wave's 16 (permuted) weight rows into LDS once; reused all 512 steps.
  {
    const unsigned short* Wd = Wcat + (size_t)dir * G4H * Ktot + (size_t)n0 * Ktot;
    const int kc = Ktot >> 3;
    for (int c = lane; c < 16 * kc; c += 32) {
      int n = c / kc, k8 = (c - n * kc) << 3;
      v8bf v = *(const v8bf*)(Wd + (size_t)n * Ktot + k8);
      *(v8bf*)(myB + n * ldstr + k8) = v;
    }
  }
  // Zero recurrent h state (c state lives in registers).
  {
    unsigned gt = blockIdx.x * blockDim.x + tid;
    for (unsigned i = gt; i < 2u * B_ * H_; i += gthreads) h_bf[i] = 0;
  }
  grid_barrier(cnt, gen, gridDim.x);

  const float biasv = bias[dir * G4H + n0 + mrow];
  const int nk  = Ktot >> 5;   // 32-wide K steps
  const int nkx = Ein  >> 5;   // K steps sourced from x_t
  const unsigned short* hrow_rd = h_bf + (size_t)dir * B_ * H_;
  unsigned short*       hrow_wr = h_bf + (size_t)dir * B_ * H_;

  // Cross-lane exchange pattern: lane c = gate*4 + jj within each 16-lane half.
  const int half  = lane & 16;
  const int jj    = lane & 3;
  const int src_i = half | jj;
  const int src_f = half | (4 + jj);
  const int src_g = half | (8 + jj);
  const int src_o = half | (12 + jj);
  const int jcol  = (strip << 2) + jj;             // hidden unit index in [0,512)
  const bool storer = ((lane & 12) == 0);          // lanes 0-3 and 16-19

  float creg[32];                                   // c state: rows (m,r) x my jcol
#pragma unroll
  for (int i = 0; i < 32; ++i) creg[i] = 0.f;

  for (int t = 0; t < S_; ++t) {
    const int tdir = dir ? (S_ - 1 - t) : t;
    v8f acc[4] = {};
    for (int kk = 0; kk < nk; ++kk) {
      const int kbase = kk << 5;
      const __bf16* bp = myB + mrow * ldstr + kbase + hi8;
      v16bf bmat = mk16(*(const v8bf*)bp, *(const v8bf*)(bp + 16));
      const bool useX = (kk < nkx);
#pragma unroll
      for (int m = 0; m < 4; ++m) {
        const int row = (m << 4) + mrow;
        const unsigned short* ap;
        if (useX) ap = x_in + ((size_t)row * S_ + tdir) * Ein + kbase + hi8;
        else      ap = hrow_rd + (size_t)row * H_ + (kbase - Ein) + hi8;
        v16bf amat = mk16(*(const v8bf*)ap, *(const v8bf*)(ap + 16));
        acc[m] = __builtin_amdgcn_wmma_f32_16x16x32_bf16(
            false, amat, false, bmat, (short)0, acc[m], false, false);
      }
    }
    // Prefetch next timestep's activations.
    if (t + 1 < S_) {
      const int tn = dir ? (S_ - 2 - t) : (t + 1);
      __builtin_prefetch(x_in + ((size_t)mrow * S_ + tn) * Ein, 0, 1);
    }
    // In-wave LSTM cell update: exchange i/f/g/o across lanes, c stays in VGPRs.
#pragma unroll
    for (int m = 0; m < 4; ++m) {
#pragma unroll
      for (int r = 0; r < 8; ++r) {
        const float v  = acc[m][r] + biasv;          // my gate value (pre-activation)
        const float iv = __shfl(v, src_i, 32);
        const float fv = __shfl(v, src_f, 32);
        const float gv = __shfl(v, src_g, 32);
        const float ov = __shfl(v, src_o, 32);
        const float cn = sigmoidf_(fv) * creg[m * 8 + r] + sigmoidf_(iv) * tanhf(gv);
        creg[m * 8 + r] = cn;
        const float hv = sigmoidf_(ov) * tanhf(cn);
        if (storer) {
          const int row = (m << 4) + r + hi8;        // batch row (C layout M index)
          const unsigned short hb = f2bf(hv);
          hrow_wr[(size_t)row * H_ + jcol] = hb;
          out[((size_t)row * S_ + tdir) * HD_ + dir * H_ + jcol] = hb;
        }
      }
    }
    grid_barrier(cnt, gen, gridDim.x);   // publish h before anyone reads it
  }
}

// ---------------- classifier: logits = out1 @ cls_w^T + cls_b (T=9) -------------
__global__ void cls_kernel(const unsigned short* __restrict__ out1, const float* __restrict__ cls_w,
                           const float* __restrict__ cls_b, float* __restrict__ logits) {
  int idx = blockIdx.x * blockDim.x + threadIdx.x;
  if (idx >= B_ * S_ * T_) return;
  const int tag = idx % T_;
  const int row = idx / T_;
  const unsigned short* xr = out1 + (size_t)row * HD_;
  const float* wr = cls_w + (size_t)tag * HD_;
  float s = cls_b[tag];
#pragma unroll 8
  for (int k = 0; k < HD_; ++k) s += bf2f(xr[k]) * wr[k];
  logits[(size_t)row * T_ + tag] = s;
}

// ---------------- CRF: loss forward + Viterbi decode, one wave per batch row ----
__global__ __launch_bounds__(32) void crf_kernel(
    const float* __restrict__ logits, const int* __restrict__ labels,
    const int* __restrict__ valid_lens, const float* __restrict__ trans,
    const float* __restrict__ start_t, const float* __restrict__ end_t,
    float* __restrict__ out_tags, float* __restrict__ llh, int* __restrict__ hist)
{
  const int b = blockIdx.x, lane = threadIdx.x;
  const float* em = logits + (size_t)b * S_ * T_;
  const int vl = valid_lens[b];
  const bool act = lane < T_;
  float tcol[T_];
#pragma unroll
  for (int i = 0; i < T_; ++i) tcol[i] = trans[i * T_ + (act ? lane : 0)];
  float alpha = act ? (start_t[lane] + em[lane]) : -1e30f;
  float score = alpha;
  int* hb = hist + (size_t)b * S_ * T_;

  for (int t = 1; t < S_; ++t) {
    float a_i[T_], s_i[T_];
#pragma unroll
    for (int i = 0; i < T_; ++i) { a_i[i] = __shfl(alpha, i, 32); s_i[i] = __shfl(score, i, 32); }
    float m = -1e30f;
#pragma unroll
    for (int i = 0; i < T_; ++i) m = fmaxf(m, a_i[i] + tcol[i]);
    float s = 0.f;
#pragma unroll
    for (int i = 0; i < T_; ++i) s += expf(a_i[i] + tcol[i] - m);
    const float emt = act ? em[t * T_ + lane] : 0.f;
    const float nxt = m + logf(s) + emt;
    float best = -1e30f; int barg = 0;
#pragma unroll
    for (int i = 0; i < T_; ++i) { float v = s_i[i] + tcol[i]; if (v > best) { best = v; barg = i; } }
    const float snew = best + emt;
    const bool mt = (t < vl);
    if (act) {
      hb[t * T_ + lane] = barg;
      if (mt) { alpha = nxt; score = snew; }
    }
  }

  float a_f[T_], s_f[T_];
#pragma unroll
  for (int i = 0; i < T_; ++i) { a_f[i] = __shfl(alpha, i, 32); s_f[i] = __shfl(score, i, 32); }

  if (lane == 0) {
    float m = -1e30f;
    for (int i = 0; i < T_; ++i) m = fmaxf(m, a_f[i] + end_t[i]);
    float s = 0.f;
    for (int i = 0; i < T_; ++i) s += expf(a_f[i] + end_t[i] - m);
    const float logZ = m + logf(s);
    float best = -1e30f; int bl = 0;
    for (int i = 0; i < T_; ++i) { float v = s_f[i] + end_t[i]; if (v > best) { best = v; bl = i; } }

    const int* lab = labels + (size_t)b * S_;
    float num = start_t[lab[0]] + em[lab[0]];
    int prevl = lab[0];
    for (int t = 1; t < S_; ++t) {
      if (t < vl) num += em[t * T_ + lab[t]] + trans[prevl * T_ + lab[t]];
      prevl = lab[t];
    }
    num += end_t[lab[vl - 1]];
    llh[b] = num - logZ;

    int tag = bl;
    out_tags[(size_t)b * S_ + (S_ - 1)] = (float)tag;
    for (int k = S_ - 2; k >= 0; --k) {
      const int prev = hb[(k + 1) * T_ + tag];
      if ((k + 1) < vl) tag = prev;
      out_tags[(size_t)b * S_ + k] = (float)tag;
    }
  }
}

__global__ void loss_kernel(const float* __restrict__ llh, float* __restrict__ out_loss) {
  if (threadIdx.x == 0) {
    float s = 0.f;
    for (int i = 0; i < B_; ++i) s += llh[i];
    out_loss[0] = -s / (float)B_;
  }
}

// ---------------- host side ------------------------------------------------------
extern "C" void kernel_launch(void* const* d_in, const int* in_sizes, int n_in,
                              void* d_out, int out_size, void* d_ws, size_t ws_size,
                              hipStream_t stream) {
  (void)in_sizes; (void)n_in; (void)out_size; (void)ws_size;
  const int*   input_ids  = (const int*)d_in[0];
  const int*   valid_lens = (const int*)d_in[1];
  const int*   labels     = (const int*)d_in[2];
  const float* emb        = (const float*)d_in[3];
  const float* w_ih0      = (const float*)d_in[4];
  const float* w_hh0      = (const float*)d_in[5];
  const float* b_ih0      = (const float*)d_in[6];
  const float* b_hh0      = (const float*)d_in[7];
  const float* w_ih1      = (const float*)d_in[8];
  const float* w_hh1      = (const float*)d_in[9];
  const float* b_ih1      = (const float*)d_in[10];
  const float* b_hh1      = (const float*)d_in[11];
  const float* cls_w      = (const float*)d_in[12];
  const float* cls_b      = (const float*)d_in[13];
  const float* trans      = (const float*)d_in[14];
  const float* start_t    = (const float*)d_in[15];
  const float* end_t      = (const float*)d_in[16];

  char* ws = (char*)d_ws;
  size_t off = 0;
  auto alloc = [&](size_t bytes) -> void* {
    void* p = ws + off;
    off = (off + bytes + 255) & ~(size_t)255;
    return p;
  };
  unsigned*       cnt   = (unsigned*)alloc(8);
  unsigned*       gen   = cnt + 1;
  unsigned short* x_bf  = (unsigned short*)alloc((size_t)B_ * S_ * E_ * 2);
  unsigned short* W0    = (unsigned short*)alloc((size_t)2 * G4H * (E_ + H_) * 2);
  float*          bias0 = (float*)alloc((size_t)2 * G4H * 4);
  unsigned short* W1    = (unsigned short*)alloc((size_t)2 * G4H * (HD_ + H_) * 2);
  float*          bias1 = (float*)alloc((size_t)2 * G4H * 4);
  unsigned short* out0  = (unsigned short*)alloc((size_t)B_ * S_ * HD_ * 2);
  unsigned short* out1  = (unsigned short*)alloc((size_t)B_ * S_ * HD_ * 2);
  unsigned short* h_bf  = (unsigned short*)alloc((size_t)2 * B_ * H_ * 2);
  float*          logit = (float*)alloc((size_t)B_ * S_ * T_ * 4);
  int*            hist  = (int*)alloc((size_t)B_ * S_ * T_ * 4);
  float*          llh   = (float*)alloc((size_t)B_ * 4);

  init_kernel<<<1, 64, 0, stream>>>(cnt, gen);

  {
    size_t n = (size_t)B_ * S_ * E_;
    embed_kernel<<<(unsigned)((n + 255) / 256), 256, 0, stream>>>(input_ids, emb, x_bf);
  }
  prepack_kernel<<<4096, 256, 0, stream>>>(w_ih0, w_hh0, b_ih0, b_hh0, W0, bias0, E_);
  prepack_kernel<<<4096, 256, 0, stream>>>(w_ih1, w_hh1, b_ih1, b_hh1, W1, bias1, HD_);

  const int shmem0 = 4 * 16 * (E_  + H_ + 8) * 2;   //  99,328 B
  const int shmem1 = 4 * 16 * (HD_ + H_ + 8) * 2;   // 197,632 B (within 320KB WGP LDS)
  lstm_layer_kernel<<<64, 128, shmem0, stream>>>(x_bf, W0, bias0, out0, h_bf, cnt, gen, E_);
  lstm_layer_kernel<<<64, 128, shmem1, stream>>>(out0, W1, bias1, out1, h_bf, cnt, gen, HD_);

  {
    int n = B_ * S_ * T_;
    cls_kernel<<<(n + 255) / 256, 256, 0, stream>>>(out1, cls_w, cls_b, logit);
  }
  crf_kernel<<<B_, 32, 0, stream>>>(logit, labels, valid_lens, trans, start_t, end_t,
                                    (float*)d_out, llh, hist);
  loss_kernel<<<1, 32, 0, stream>>>(llh, ((float*)d_out) + (size_t)B_ * S_);
}
